// TalonGPT_18159121728205
// MI455X (gfx1250) — matrix operand order repaired and checked by reading, not compiled, not run on hardware
//
#include <hip/hip_runtime.h>
#include <hip/hip_bf16.h>
#include <math.h>

// GPT-2 small-ish config from the reference
#define GPT_V 50257
#define GPT_T 1024
#define GPT_D 768
#define GPT_H 12
#define GPT_L 6
#define GPT_B 2
#define GPT_S 1024
#define GPT_N (GPT_B * GPT_S)   // 2048 tokens
#define GPT_HD 64               // head dim

typedef __attribute__((ext_vector_type(16))) __bf16 v16bf;
typedef __attribute__((ext_vector_type(8)))  float  v8f;
typedef __attribute__((ext_vector_type(4)))  unsigned int gpt_u32x4;
typedef __attribute__((ext_vector_type(8)))  int          gpt_i32x8;
typedef __attribute__((ext_vector_type(4)))  int          gpt_i32x4;

// ---------------------------------------------------------------------------
// WMMA fragment layouts (CDNA5 ISA 7.12.2):
// A (16x32 bf16, MxK): lane row = lane&15, half = lane>>4.
//   elems 0..7  <- K = k0 + 8*half + {0..7}
//   elems 8..15 <- K = k0 + 16 + 8*half + {0..7}
// B (32x16 bf16, KxN): lane col = lane&15; same K order as A
// C/D (16x16 f32): VGPR r, lane l -> row = r + 8*(l>>4), col = l&15
// ---------------------------------------------------------------------------
static __device__ __forceinline__ v16bf gpt_load_a_frag(const float* __restrict__ rowp,
                                                        int k0, int half) {
  const float* p = rowp + k0 + 8 * half;
  v16bf f;
#pragma unroll
  for (int i = 0; i < 8; ++i) f[i] = (__bf16)p[i];
#pragma unroll
  for (int i = 0; i < 8; ++i) f[8 + i] = (__bf16)p[16 + i];
  return f;
}

static __device__ __forceinline__ v16bf gpt_load_b_frag(const float* __restrict__ rowp,
                                                        int k0, int half) {
  const float* p = rowp + k0 + 8 * half;
  v16bf f;
#pragma unroll
  for (int i = 0; i < 8; ++i) { f[i] = (__bf16)p[i]; f[8 + i] = (__bf16)p[16 + i]; }
  return f;
}

static __device__ __forceinline__ v8f gpt_wmma(v16bf a, v16bf b, v8f c) {
  return __builtin_amdgcn_wmma_f32_16x16x32_bf16(false, a, false, b, (short)0, c,
                                                 false, false);
}

// ---------------------------------------------------------------------------
// TDM: issue a 2D tensor_load_to_lds.
//   tile = tileW x tileH elements (4B each), row stride = strideElems,
//   LDS rows padded by 2 DWORDs every 32 DWORDs (row stride 34 floats:
//   conflict-free bank mapping for the fragment gathers below).
//   Rows >= tensorH are zero-filled by TDM OOB handling (ragged vocab edge).
// D# bit layout per CDNA5 ISA 8.3/8.4.  6-arg builtin form on this toolchain.
// ---------------------------------------------------------------------------
static __device__ __forceinline__ void gpt_tdm_load_2d(
    unsigned ldsByteOff, const float* gaddr, unsigned tileW, unsigned tileH,
    unsigned long long tensorW, unsigned long long tensorH,
    unsigned long long strideElems) {
  const unsigned long long ga = (unsigned long long)(uintptr_t)gaddr;
  gpt_u32x4 g0;
  g0[0] = 1u;                                   // count=1, user descriptor
  g0[1] = ldsByteOff;                           // lds_addr
  g0[2] = (unsigned)(ga & 0xFFFFFFFFull);       // global_addr[31:0]
  g0[3] = (unsigned)((ga >> 32) & 0x01FFFFFFull) | (2u << 30);  // addr[56:32]|type=2

  // group1, 256 bits as four u64 halves
  const unsigned long long q0 =
      (2ull << 16)                               // data_size = 4B
    | (1ull << 20)                               // pad_enable
    | (4ull << 22)                               // pad_interval: 32 DWORDs
    | (1ull << 25)                               // pad_amount: 2 DWORDs
    | ((tensorW & 0xFFFFull) << 48);             // tensor_dim0[15:0]
  const unsigned long long q1 =
      ((tensorW >> 16) & 0xFFFFull)              // tensor_dim0[31:16]
    | ((tensorH & 0xFFFFFFFFull) << 16)          // tensor_dim1
    | ((unsigned long long)(tileW & 0xFFFFu) << 48);  // tile_dim0
  const unsigned long long q2 =
      (unsigned long long)(tileH & 0xFFFFu)      // tile_dim1 (tile_dim2 = 0)
    | ((strideElems & 0xFFFFFFFFull) << 32);     // tensor_dim0_stride[31:0]
  const unsigned long long q3 =
      (strideElems >> 32) & 0xFFFFull;           // tensor_dim0_stride[47:32]

  gpt_i32x8 g1;
  g1[0] = (int)q0; g1[1] = (int)(q0 >> 32);
  g1[2] = (int)q1; g1[3] = (int)(q1 >> 32);
  g1[4] = (int)q2; g1[5] = (int)(q2 >> 32);
  g1[6] = (int)q3; g1[7] = (int)(q3 >> 32);
  const gpt_i32x4 gz4 = {0, 0, 0, 0};
  const gpt_i32x8 gz8 = {0, 0, 0, 0, 0, 0, 0, 0};
  __builtin_amdgcn_tensor_load_to_lds(g0, g1, gz4, gz4, gz8, 0);
}

// ---------------------------------------------------------------------------
// GEMM: out[M,Nc] = A[M,K] @ W[Nc,K]^T + bias (+ GELU) (+ residual)
// 256-thread block (8 waves) computes a 64x128 tile; A/B tiles staged in LDS
// by the Tensor Data Mover, double-buffered; each wave owns a 32x32 subtile
// (4 WMMAs per 32-wide K step). K % 32 == 0, M % 64 == 0, Nc ragged OK.
// ---------------------------------------------------------------------------
#define GEMM_TM 64
#define GEMM_TN 128
#define GEMM_TK 32
#define GEMM_LDSW 34                         // 32 data + 2 pad DWORDs per row
#define GEMM_AOFF 0                          // A tile: 64 rows
#define GEMM_BOFF (GEMM_TM * GEMM_LDSW)      // B tile: 128 rows
#define GEMM_STAGE ((GEMM_TM + GEMM_TN) * GEMM_LDSW)  // floats per stage

static __device__ __forceinline__ void gpt_store_tile(
    v8f acc, float* __restrict__ out, int Nc, int mBase, int nBase,
    const float* __restrict__ bias, const float* __restrict__ residual,
    int fuseGelu, int half, int l15) {
  const int col = nBase + l15;
  if (col >= Nc) return;
  const float bcol = bias ? bias[col] : 0.0f;
#pragma unroll
  for (int r = 0; r < 8; ++r) {
    const int row = mBase + r + 8 * half;
    float v = acc[r] + bcol;
    if (fuseGelu) v = 0.5f * v * (1.0f + erff(v * 0.70710678118654752f));
    const size_t o = (size_t)row * Nc + col;
    if (residual) v += residual[o];
    out[o] = v;
  }
}

__global__ void __launch_bounds__(256) gpt_gemm_wmma(
    const float* __restrict__ A, const float* __restrict__ W,
    const float* __restrict__ bias, const float* __restrict__ residual,
    float* __restrict__ out, int M, int Nc, int K, int fuseGelu) {
  __shared__ float lds[2 * GEMM_STAGE];

  const int tid  = threadIdx.x;
  const int lane = tid & 31;
  const int wave = tid >> 5;         // 0..7
  const int half = lane >> 4;
  const int l15  = lane & 15;
  const int wm   = wave >> 2;        // 0..1 : 32-row band inside the 64 rows
  const int wn   = wave & 3;         // 0..3 : 32-col band inside the 128 cols

  const int n0 = blockIdx.x * GEMM_TN;
  const int m0 = blockIdx.y * GEMM_TM;

  // initial stage 0 copy via TDM (wave 0 issues; EXEC-independent DMA)
  if (wave == 0) {
    gpt_tdm_load_2d(GEMM_AOFF * 4u, A + (size_t)m0 * K, GEMM_TK, GEMM_TM,
                    (unsigned long long)K, (unsigned long long)(M - m0),
                    (unsigned long long)K);
    gpt_tdm_load_2d(GEMM_BOFF * 4u, W + (size_t)n0 * K, GEMM_TK, GEMM_TN,
                    (unsigned long long)K, (unsigned long long)(Nc - n0),
                    (unsigned long long)K);
    __builtin_amdgcn_s_wait_tensorcnt(0);
  }
  __syncthreads();

  v8f acc00 = {}, acc01 = {}, acc10 = {}, acc11 = {};

  int s = 0;
  for (int k0 = 0; k0 < K; k0 += GEMM_TK, s ^= 1) {
    // prefetch next stage into the other buffer while this one is consumed
    const int kn = k0 + GEMM_TK;
    if (wave == 0 && kn < K) {
      const unsigned nb = (unsigned)((s ^ 1) * GEMM_STAGE * 4);
      gpt_tdm_load_2d(nb + GEMM_AOFF * 4u, A + (size_t)m0 * K + kn, GEMM_TK,
                      GEMM_TM, (unsigned long long)(K - kn),
                      (unsigned long long)(M - m0), (unsigned long long)K);
      gpt_tdm_load_2d(nb + GEMM_BOFF * 4u, W + (size_t)n0 * K + kn, GEMM_TK,
                      GEMM_TN, (unsigned long long)(K - kn),
                      (unsigned long long)(Nc - n0), (unsigned long long)K);
    }

    // The TDM builtin has no pointer operand, so clang cannot see that it
    // writes lds[]. Escape the array's address into an opaque asm with a
    // memory clobber so the fragment ds_loads below actually read LDS.
    asm volatile("" : : "v"(&lds[0]) : "memory");

    // build fragments from LDS (fp32 -> bf16 in-register)
    const int sb = s * GEMM_STAGE;
    v16bf a0, a1, b0, b1;
    {
      const int ra0 = wm * 32 + l15;
      const int ra1 = ra0 + 16;
      const float* pa0 = &lds[sb + GEMM_AOFF + ra0 * GEMM_LDSW + 8 * half];
      const float* pa1 = &lds[sb + GEMM_AOFF + ra1 * GEMM_LDSW + 8 * half];
#pragma unroll
      for (int i = 0; i < 8; ++i) { a0[i] = (__bf16)pa0[i]; a0[8+i] = (__bf16)pa0[16+i]; }
#pragma unroll
      for (int i = 0; i < 8; ++i) { a1[i] = (__bf16)pa1[i]; a1[8+i] = (__bf16)pa1[16+i]; }
      const int rb0 = wn * 32 + l15;
      const int rb1 = rb0 + 16;
      const float* pb0 = &lds[sb + GEMM_BOFF + rb0 * GEMM_LDSW + 8 * half];
      const float* pb1 = &lds[sb + GEMM_BOFF + rb1 * GEMM_LDSW + 8 * half];
#pragma unroll
      for (int i = 0; i < 8; ++i) { b0[i] = (__bf16)pb0[i]; b0[8+i] = (__bf16)pb0[16+i]; }
#pragma unroll
      for (int i = 0; i < 8; ++i) { b1[i] = (__bf16)pb1[i]; b1[8+i] = (__bf16)pb1[16+i]; }
    }

    acc00 = gpt_wmma(a0, b0, acc00);
    acc01 = gpt_wmma(a0, b1, acc01);
    acc10 = gpt_wmma(a1, b0, acc10);
    acc11 = gpt_wmma(a1, b1, acc11);

    if (wave == 0) __builtin_amdgcn_s_wait_tensorcnt(0);
    __syncthreads();
  }

  const int mB = m0 + wm * 32;
  const int nB = n0 + wn * 32;
  gpt_store_tile(acc00, out, Nc, mB,      nB,      bias, residual, fuseGelu, half, l15);
  gpt_store_tile(acc01, out, Nc, mB,      nB + 16, bias, residual, fuseGelu, half, l15);
  gpt_store_tile(acc10, out, Nc, mB + 16, nB,      bias, residual, fuseGelu, half, l15);
  gpt_store_tile(acc11, out, Nc, mB + 16, nB + 16, bias, residual, fuseGelu, half, l15);
}

// ---------------------------------------------------------------------------
// Flash attention (wave32): one wave per (b, h, 16-query tile).
// ---------------------------------------------------------------------------
__global__ void __launch_bounds__(32) gpt_attn_wmma(const float* __restrict__ qkv,
                                                    float* __restrict__ out) {
  const int qt = blockIdx.x;
  const int h  = blockIdx.y;
  const int b  = blockIdx.z;
  const int lane = threadIdx.x & 31;
  const int half = lane >> 4;
  const int l15  = lane & 15;

  const int ld = 3 * GPT_D;
  const float* qbase = qkv + (size_t)(b * GPT_S) * ld + h * GPT_HD;
  const float* kbase = qbase + GPT_D;
  const float* vbase = qbase + 2 * GPT_D;

  const float* qrow = qbase + (size_t)(qt * 16 + l15) * ld;
  const v16bf qa0 = gpt_load_a_frag(qrow, 0, half);
  const v16bf qa1 = gpt_load_a_frag(qrow, 32, half);

  float m_i[8], l_i[8];
#pragma unroll
  for (int r = 0; r < 8; ++r) { m_i[r] = -3.0e38f; l_i[r] = 0.0f; }
  const v8f vzero = {};
  v8f oacc[4];
#pragma unroll
  for (int c = 0; c < 4; ++c) oacc[c] = vzero;

  __shared__ float ptile[16 * 17];

  for (int kt = 0; kt <= qt; ++kt) {
    v8f sc = vzero;
    {
      const float* kr = kbase + (size_t)(kt * 16 + l15) * ld;
      v16bf kb0 = gpt_load_b_frag(kr, 0, half);
      v16bf kb1 = gpt_load_b_frag(kr, 32, half);
      sc = gpt_wmma(qa0, kb0, sc);
      sc = gpt_wmma(qa1, kb1, sc);
    }

    const bool diag = (kt == qt);
#pragma unroll
    for (int r = 0; r < 8; ++r) {
      const int m = r + 8 * half;
      float sv = sc[r] * 0.125f;
      if (diag && (l15 > m)) sv = -3.0e38f;
      float mx = sv;
      mx = fmaxf(mx, __shfl_xor(mx, 1, 32));
      mx = fmaxf(mx, __shfl_xor(mx, 2, 32));
      mx = fmaxf(mx, __shfl_xor(mx, 4, 32));
      mx = fmaxf(mx, __shfl_xor(mx, 8, 32));
      const float mnew = fmaxf(m_i[r], mx);
      const float corr = __expf(m_i[r] - mnew);
      const float pv   = __expf(sv - mnew);
      float rs = pv;
      rs += __shfl_xor(rs, 1, 32);
      rs += __shfl_xor(rs, 2, 32);
      rs += __shfl_xor(rs, 4, 32);
      rs += __shfl_xor(rs, 8, 32);
      l_i[r] = l_i[r] * corr + rs;
      m_i[r] = mnew;
#pragma unroll
      for (int c = 0; c < 4; ++c) oacc[c][r] *= corr;
      ptile[m * 17 + l15] = pv;
    }
    __syncthreads();

    v16bf pa;
#pragma unroll
    for (int e = 0; e < 8; ++e) pa[e] = (__bf16)ptile[l15 * 17 + e + 8 * half];
#pragma unroll
    for (int e = 8; e < 16; ++e) pa[e] = (__bf16)0.0f;
    __syncthreads();

#pragma unroll
    for (int c = 0; c < 4; ++c) {
      v16bf vb;
#pragma unroll
      for (int e = 0; e < 16; ++e) {
        int krow = kt * 16 + (e < 8 ? e + 8 * half : 8 + 8 * half + e);
        krow = min(krow, GPT_S - 1);
        vb[e] = (__bf16)vbase[(size_t)krow * ld + c * 16 + l15];
      }
      oacc[c] = gpt_wmma(pa, vb, oacc[c]);
    }
  }

#pragma unroll
  for (int r = 0; r < 8; ++r) {
    const float inv = 1.0f / l_i[r];
    const int row = b * GPT_S + qt * 16 + r + 8 * half;
#pragma unroll
    for (int c = 0; c < 4; ++c)
      out[(size_t)row * GPT_D + h * GPT_HD + c * 16 + l15] = oacc[c][r] * inv;
  }
}

// ---------------------------------------------------------------------------
// Block-wide reductions (blockDim == 256 -> 8 wave32s)
// ---------------------------------------------------------------------------
static __device__ float gpt_block_sum_256(float v) {
  __shared__ float sm[8];
#pragma unroll
  for (int mk = 16; mk >= 1; mk >>= 1) v += __shfl_xor(v, mk, 32);
  if ((threadIdx.x & 31) == 0) sm[threadIdx.x >> 5] = v;
  __syncthreads();
  float a = (threadIdx.x < 8) ? sm[threadIdx.x] : 0.0f;
#pragma unroll
  for (int mk = 4; mk >= 1; mk >>= 1) a += __shfl_xor(a, mk, 32);
  if (threadIdx.x == 0) sm[0] = a;
  __syncthreads();
  const float r = sm[0];
  __syncthreads();
  return r;
}

static __device__ float gpt_block_max_256(float v) {
  __shared__ float sm[8];
#pragma unroll
  for (int mk = 16; mk >= 1; mk >>= 1) v = fmaxf(v, __shfl_xor(v, mk, 32));
  if ((threadIdx.x & 31) == 0) sm[threadIdx.x >> 5] = v;
  __syncthreads();
  float a = (threadIdx.x < 8) ? sm[threadIdx.x] : -3.0e38f;
#pragma unroll
  for (int mk = 4; mk >= 1; mk >>= 1) a = fmaxf(a, __shfl_xor(a, mk, 32));
  if (threadIdx.x == 0) sm[0] = a;
  __syncthreads();
  const float r = sm[0];
  __syncthreads();
  return r;
}

// ---------------------------------------------------------------------------
// LayerNorm: one 256-thread block per token row.
// ---------------------------------------------------------------------------
__global__ void __launch_bounds__(256) gpt_layernorm(const float* __restrict__ x,
                                                     const float* __restrict__ g,
                                                     const float* __restrict__ be,
                                                     float* __restrict__ o) {
  const int row = blockIdx.x;
  const float* xr = x + (size_t)row * GPT_D;
  float s = 0.0f, s2 = 0.0f;
  for (int i = threadIdx.x; i < GPT_D; i += blockDim.x) {
    const float v = xr[i];
    s += v;
    s2 += v * v;
  }
  const float tot  = gpt_block_sum_256(s);
  const float tot2 = gpt_block_sum_256(s2);
  const float mean = tot * (1.0f / GPT_D);
  const float var  = tot2 * (1.0f / GPT_D) - mean * mean;
  const float rstd = rsqrtf(var + 1e-5f);
  float* orow = o + (size_t)row * GPT_D;
  for (int i = threadIdx.x; i < GPT_D; i += blockDim.x)
    orow[i] = (xr[i] - mean) * rstd * g[i] + be[i];
}

// ---------------------------------------------------------------------------
// Embedding: x[n,d] = wte[idx[n], d] + wpe[n % S, d]
// ---------------------------------------------------------------------------
__global__ void gpt_embed(const int* __restrict__ idx, const float* __restrict__ wte,
                          const float* __restrict__ wpe, float* __restrict__ x) {
  const size_t i = (size_t)blockIdx.x * blockDim.x + threadIdx.x;
  if (i >= (size_t)GPT_N * GPT_D) return;
  const int row = (int)(i / GPT_D);
  const int d   = (int)(i % GPT_D);
  const int tok = idx[row];
  const int pos = row % GPT_S;
  x[i] = wte[(size_t)tok * GPT_D + d] + wpe[(size_t)pos * GPT_D + d];
}

// ---------------------------------------------------------------------------
// Loss: per-row  lse - logit[target], then mean.
// ---------------------------------------------------------------------------
__global__ void __launch_bounds__(256) gpt_loss_rows(const float* __restrict__ logits,
                                                     const int* __restrict__ targets,
                                                     float* __restrict__ rowloss) {
  const int row = blockIdx.x;
  const float* lr = logits + (size_t)row * GPT_V;
  float mx = -3.0e38f;
  for (int i = threadIdx.x; i < GPT_V; i += blockDim.x) mx = fmaxf(mx, lr[i]);
  mx = gpt_block_max_256(mx);
  float s = 0.0f;
  for (int i = threadIdx.x; i < GPT_V; i += blockDim.x) s += __expf(lr[i] - mx);
  const float tot = gpt_block_sum_256(s);
  if (threadIdx.x == 0)
    rowloss[row] = (mx + logf(tot)) - lr[targets[row]];
}

__global__ void __launch_bounds__(256) gpt_loss_mean(const float* __restrict__ rowloss,
                                                     float* __restrict__ outLoss) {
  float s = 0.0f;
  for (int i = threadIdx.x; i < GPT_N; i += blockDim.x) s += rowloss[i];
  const float tot = gpt_block_sum_256(s);
  if (threadIdx.x == 0) outLoss[0] = tot * (1.0f / GPT_N);
}

// ---------------------------------------------------------------------------
// Orchestration
// ---------------------------------------------------------------------------
extern "C" void kernel_launch(void* const* d_in, const int* in_sizes, int n_in,
                              void* d_out, int out_size, void* d_ws, size_t ws_size,
                              hipStream_t stream) {
  (void)in_sizes; (void)n_in; (void)out_size; (void)ws_size;

  const int*   idx     = (const int*)d_in[0];
  const int*   targets = (const int*)d_in[1];
  const float* wte     = (const float*)d_in[2];
  const float* wpe     = (const float*)d_in[3];
  const float* ln1_g   = (const float*)d_in[4];
  const float* ln1_b   = (const float*)d_in[5];
  const float* qkv_w   = (const float*)d_in[6];
  const float* qkv_b   = (const float*)d_in[7];
  const float* proj_w  = (const float*)d_in[8];
  const float* proj_b  = (const float*)d_in[9];
  const float* ln2_g   = (const float*)d_in[10];
  const float* ln2_b   = (const float*)d_in[11];
  const float* ff1_w   = (const float*)d_in[12];
  const float* ff1_b   = (const float*)d_in[13];
  const float* ff2_w   = (const float*)d_in[14];
  const float* ff2_b   = (const float*)d_in[15];
  const float* lnf_g   = (const float*)d_in[16];
  const float* lnf_b   = (const float*)d_in[17];

  float* logits = (float*)d_out;
  float* loss   = logits + (size_t)GPT_N * GPT_V;

  float* ws   = (float*)d_ws;
  float* x    = ws;                                  // [N, D]
  float* t1   = x    + (size_t)GPT_N * GPT_D;        // [N, D]
  float* qkvb = t1   + (size_t)GPT_N * GPT_D;        // [N, 3D]
  float* ao   = qkvb + (size_t)GPT_N * 3 * GPT_D;    // [N, D]
  float* ffh  = ao   + (size_t)GPT_N * GPT_D;        // [N, 4D]
  float* rl   = ffh  + (size_t)GPT_N * 4 * GPT_D;    // [N]

  gpt_embed<<<dim3((GPT_N * GPT_D + 255) / 256), dim3(256), 0, stream>>>(idx, wte, wpe, x);

  for (int l = 0; l < GPT_L; ++l) {
    gpt_layernorm<<<GPT_N, 256, 0, stream>>>(x, ln1_g + (size_t)l * GPT_D,
                                             ln1_b + (size_t)l * GPT_D, t1);
    gpt_gemm_wmma<<<dim3(3 * GPT_D / GEMM_TN, GPT_N / GEMM_TM), 256, 0, stream>>>(
        t1, qkv_w + (size_t)l * 3 * GPT_D * GPT_D, qkv_b + (size_t)l * 3 * GPT_D,
        nullptr, qkvb, GPT_N, 3 * GPT_D, GPT_D, 0);
    gpt_attn_wmma<<<dim3(GPT_S / 16, GPT_H, GPT_B), 32, 0, stream>>>(qkvb, ao);
    gpt_gemm_wmma<<<dim3(GPT_D / GEMM_TN, GPT_N / GEMM_TM), 256, 0, stream>>>(
        ao, proj_w + (size_t)l * GPT_D * GPT_D, proj_b + (size_t)l * GPT_D,
        x, x, GPT_N, GPT_D, GPT_D, 0);
    gpt_layernorm<<<GPT_N, 256, 0, stream>>>(x, ln2_g + (size_t)l * GPT_D,
                                             ln2_b + (size_t)l * GPT_D, t1);
    gpt_gemm_wmma<<<dim3(4 * GPT_D / GEMM_TN, GPT_N / GEMM_TM), 256, 0, stream>>>(
        t1, ff1_w + (size_t)l * 4 * GPT_D * GPT_D, ff1_b + (size_t)l * 4 * GPT_D,
        nullptr, ffh, GPT_N, 4 * GPT_D, GPT_D, 1 /*gelu*/);
    gpt_gemm_wmma<<<dim3(GPT_D / GEMM_TN, GPT_N / GEMM_TM), 256, 0, stream>>>(
        ffh, ff2_w + (size_t)l * GPT_D * 4 * GPT_D, ff2_b + (size_t)l * GPT_D,
        x, x, GPT_N, GPT_D, 4 * GPT_D, 0);
  }

  gpt_layernorm<<<GPT_N, 256, 0, stream>>>(x, lnf_g, lnf_b, t1);
  gpt_gemm_wmma<<<dim3((GPT_V + GEMM_TN - 1) / GEMM_TN, GPT_N / GEMM_TM), 256, 0, stream>>>(
      t1, wte, nullptr, nullptr, logits, GPT_N, GPT_V, GPT_D, 0);

  gpt_loss_rows<<<GPT_N, 256, 0, stream>>>(logits, targets, rl);
  gpt_loss_mean<<<1, 256, 0, stream>>>(rl, loss);
}
